// MultiHeadAttention_55027120996458
// MI455X (gfx1250) — matrix operand order repaired and checked by reading, not compiled, not run on hardware
//
#include <hip/hip_runtime.h>
#include <cstdint>
#include <cstddef>

#define HIDDEN   1024
#define NUM_HEAD 16
#define HEAD_DIM 64
#define BATCH    2
#define SEQ      2048
#define TOKENS   (BATCH * SEQ)   // 4096

typedef __attribute__((ext_vector_type(16))) __bf16 bf16x16;
typedef __attribute__((ext_vector_type(8)))  __bf16 bf16x8;
typedef __attribute__((ext_vector_type(8)))  float  f32x8;

// ---- helpers -------------------------------------------------------------

__device__ __forceinline__ unsigned short f2bf(float f) {
  __bf16 h = (__bf16)f;                      // RNE; packs to v_cvt_pk_bf16_f32
  return __builtin_bit_cast(unsigned short, h);
}

__device__ __forceinline__ bf16x8 ld8(const unsigned short* p) {
  bf16x8 v; __builtin_memcpy(&v, p, 16); return v;
}
__device__ __forceinline__ bf16x16 ld16(const unsigned short* p) {
  bf16x16 v; __builtin_memcpy(&v, p, 32); return v;
}
__device__ __forceinline__ bf16x16 cat(bf16x8 lo, bf16x8 hi) {
  bf16x16 v;
#pragma unroll
  for (int i = 0; i < 8; ++i) { v[i] = lo[i]; v[8 + i] = hi[i]; }
  return v;
}

__device__ __forceinline__ void st_out(float* p, float v) { *p = v; }
__device__ __forceinline__ void st_out(unsigned short* p, float v) { *p = f2bf(v); }

// LDS byte offset of a __shared__ object (generic AMDGCN pointer: low 32 bits)
__device__ __forceinline__ unsigned lds_off(const void* p) {
  return (unsigned)(size_t)p;
}

// Async DMA of an 8 KB tile of 64 rows x 128 B (128 threads, 4 x b128 each).
__device__ __forceinline__ void async_cp_tile128(unsigned lds_base,
                                                 const unsigned short* g_base,
                                                 size_t row_bytes, int t) {
#pragma unroll
  for (int it = 0; it < 4; ++it) {
    const unsigned o = (unsigned)((it * 128 + t) * 16);          // 0..8191
    const char* src = (const char*)g_base + (size_t)(o >> 7) * row_bytes + (o & 127u);
    asm volatile("global_load_async_to_lds_b128 %0, %1, off"
                 :: "v"(lds_base + o), "v"(src)
                 : "memory");
  }
}

// Async DMA of an 8 KB tile of 128 rows x 64 B (256 threads, 2 x b128 each).
__device__ __forceinline__ void async_cp_tile64(unsigned lds_base,
                                                const unsigned short* g_base,
                                                size_t row_bytes, int t) {
#pragma unroll
  for (int it = 0; it < 2; ++it) {
    const unsigned o = (unsigned)((it * 256 + t) * 16);          // 0..8191
    const char* src = (const char*)g_base + (size_t)(o >> 6) * row_bytes + (o & 63u);
    asm volatile("global_load_async_to_lds_b128 %0, %1, off"
                 :: "v"(lds_base + o), "v"(src)
                 : "memory");
  }
}

// ---- one-shot precision/layout prep --------------------------------------

// elementwise f32 -> bf16, 8 elements per thread
__global__ __launch_bounds__(256) void cvt_bf16_kernel(
    const float* __restrict__ src, unsigned short* __restrict__ dst) {
  const size_t i = ((size_t)blockIdx.x * 256 + threadIdx.x) * 8;
  float v[8];
  __builtin_memcpy(v, src + i, 32);
  unsigned short o[8];
#pragma unroll
  for (int j = 0; j < 8; ++j) o[j] = f2bf(v[j]);
  __builtin_memcpy(dst + i, o, 16);
}

// W[K][N] f32 -> Wt[N][K] bf16, 32x32 LDS tile transpose
__global__ __launch_bounds__(256) void transpose_cvt_kernel(
    const float* __restrict__ W, unsigned short* __restrict__ Wt,
    int K, int N) {
  __shared__ float tile[32][33];
  const int nx = blockIdx.x * 32, kx = blockIdx.y * 32;
  const int tx = threadIdx.x & 31, ty = threadIdx.x >> 5;   // 32 x 8
#pragma unroll
  for (int j = 0; j < 32; j += 8)
    tile[ty + j][tx] = W[(size_t)(kx + ty + j) * N + nx + tx];
  __syncthreads();
#pragma unroll
  for (int j = 0; j < 32; j += 8)
    Wt[(size_t)(nx + ty + j) * K + kx + tx] = f2bf(tile[tx][ty + j]);
}

// ---- GEMM: C[M,N] = A[M,K] @ Wt[N,K]^T + bias, all-bf16 tiles via async DMA
// Block: 256 threads = 8 waves; block tile 128x128; wave tile 32x64 (2x4 of 16x16).
// TRANSV: write output as [b][h][d][s] bf16 (head-transposed V for attention).

template <typename TO, bool TRANSV>
__global__ __launch_bounds__(256) void gemm_bias_kernel(
    const unsigned short* __restrict__ A,    // [M][K] bf16
    const unsigned short* __restrict__ Wt,   // [N][K] bf16 (pre-transposed)
    const float* __restrict__ bias, TO* __restrict__ C,
    int M, int N, int K, float out_scale) {
  __shared__ unsigned short As[2][128][32];  // [buf][m][k]
  __shared__ unsigned short Bs[2][128][32];  // [buf][n][k]

  const int t     = threadIdx.x;
  const int lane  = t & 31;
  const int wv    = t >> 5;
  const int waveM = wv >> 1;                 // 0..3
  const int waveN = wv & 1;                  // 0..1
  const int ln    = lane & 15;
  const int lh    = lane >> 4;

  const int blockN = blockIdx.x * 128;
  const int blockM = blockIdx.y * 128;

  auto issue_tile = [&](int buf, int kk) {
    async_cp_tile64(lds_off(&As[buf][0][0]), A + (size_t)blockM * K + kk,
                    (size_t)K * 2, t);
    async_cp_tile64(lds_off(&Bs[buf][0][0]), Wt + (size_t)blockN * K + kk,
                    (size_t)K * 2, t);
  };

  f32x8 acc[2][4];
#pragma unroll
  for (int nt = 0; nt < 4; ++nt) {
    const float bv = bias[blockN + waveN * 64 + nt * 16 + ln];
#pragma unroll
    for (int mt = 0; mt < 2; ++mt)
#pragma unroll
      for (int r = 0; r < 8; ++r) acc[mt][nt][r] = bv;
  }

  issue_tile(0, 0);
  for (int kk = 0; kk < K; kk += 32) {
    const int buf = (kk >> 5) & 1;
    if (kk + 32 < K) {
      issue_tile(buf ^ 1, kk + 32);                       // 4 ops in flight
      asm volatile("s_wait_asynccnt 0x4" ::: "memory");   // current tile done
    } else {
      asm volatile("s_wait_asynccnt 0x0" ::: "memory");
    }
    __syncthreads();

    bf16x16 afr[2];
#pragma unroll
    for (int mt = 0; mt < 2; ++mt) {
      const unsigned short* base = &As[buf][waveM * 32 + mt * 16 + ln][0];
      afr[mt] = cat(ld8(base + lh * 16), ld8(base + lh * 16 + 8));
    }
#pragma unroll
    for (int nt = 0; nt < 4; ++nt) {
      const bf16x16 bfr = ld16(&Bs[buf][waveN * 64 + nt * 16 + ln][lh * 16]);
#pragma unroll
      for (int mt = 0; mt < 2; ++mt)
        acc[mt][nt] = __builtin_amdgcn_wmma_f32_16x16x32_bf16(
            false, afr[mt], false, bfr, (short)0, acc[mt][nt], false, false);
    }
    __syncthreads();
  }

#pragma unroll
  for (int mt = 0; mt < 2; ++mt)
#pragma unroll
    for (int nt = 0; nt < 4; ++nt) {
      const int col = blockN + waveN * 64 + nt * 16 + ln;
      if (TRANSV) {
        // 8 consecutive tokens -> 8 consecutive s; one 16 B store
        const int row0 = blockM + waveM * 32 + mt * 16 + 8 * lh;
        const int bb = row0 >> 11;            // SEQ = 2048
        const int s0 = row0 & (SEQ - 1);
        const int hh = col >> 6, dd = col & (HEAD_DIM - 1);
        unsigned short tmp[8];
#pragma unroll
        for (int r = 0; r < 8; ++r) tmp[r] = f2bf(acc[mt][nt][r] * out_scale);
        unsigned short* dst = (unsigned short*)C +
            ((size_t)((bb * NUM_HEAD + hh) * HEAD_DIM + dd)) * SEQ + s0;
        __builtin_memcpy(dst, tmp, 16);
      } else {
#pragma unroll
        for (int r = 0; r < 8; ++r) {
          const int row = blockM + waveM * 32 + mt * 16 + r + 8 * lh;
          st_out(&C[(size_t)row * N + col], acc[mt][nt][r] * out_scale);
        }
      }
    }
}

// ---- Flash attention: 1 block = 4 waves = 64 queries of one (b,h) --------
// Q arrives pre-scaled by 1/sqrt(d). K tiles [key][d] and pre-transposed V
// tiles [d][key] via GLOBAL_LOAD_ASYNC_TO_LDS_B128, double-buffered.
// Softmax denominator is accumulated on the matrix unit: l = P @ ones.

__global__ __launch_bounds__(128) void flash_attn_kernel(
    const unsigned short* __restrict__ Qb,
    const unsigned short* __restrict__ Kb,
    const unsigned short* __restrict__ Vt_ws,   // [b][h][d][s] bf16
    unsigned short* __restrict__ AO) {
  __shared__ unsigned short Ks[2][64][64];     // [buf][key][d]
  __shared__ unsigned short Vt[2][64][64];     // [buf][d][key]
  __shared__ unsigned short Pl[4][16][64];     // per-wave P tile [q][key]

  const int t    = threadIdx.x;
  const int lane = t & 31;
  const int wv   = t >> 5;
  const int ln   = lane & 15;
  const int lh   = lane >> 4;

  const int qb = blockIdx.x;      // query block of 64
  const int h  = blockIdx.y;
  const int b  = blockIdx.z;

  const int    q0      = qb * 64 + wv * 16;
  const size_t rowstr  = HIDDEN;
  const size_t headoff = (size_t)h * HEAD_DIM;

  const unsigned short* kpanel = Kb + ((size_t)b * SEQ) * rowstr + headoff;
  const unsigned short* vpanel =
      Vt_ws + ((size_t)(b * NUM_HEAD + h) * HEAD_DIM) * SEQ;

  auto issue_tile = [&](int buf, int kv) {
    async_cp_tile128(lds_off(&Ks[buf][0][0]), kpanel + (size_t)kv * rowstr,
                     rowstr * 2, t);
    async_cp_tile128(lds_off(&Vt[buf][0][0]), vpanel + kv, SEQ * 2, t);
  };

  // Q fragments (16x64 split into two 16x32 A-frags), held in registers
  const unsigned short* qrow =
      Qb + ((size_t)b * SEQ + q0 + ln) * rowstr + headoff;
  const bf16x16 qa0 = cat(ld8(qrow + lh * 16),      ld8(qrow + lh * 16 + 8));
  const bf16x16 qa1 = cat(ld8(qrow + 32 + lh * 16), ld8(qrow + 40 + lh * 16));

  // all-ones B fragment: P @ ones puts each row's sum in every lane
  bf16x16 onesb;
#pragma unroll
  for (int i = 0; i < 16; ++i) onesb[i] = (__bf16)1.0f;

  float m_i[8];
  f32x8 l_acc, O[4];
#pragma unroll
  for (int r = 0; r < 8; ++r) { m_i[r] = -1e30f; l_acc[r] = 0.f; }
#pragma unroll
  for (int nt = 0; nt < 4; ++nt)
#pragma unroll
    for (int r = 0; r < 8; ++r) O[nt][r] = 0.f;

  const int ntiles = qb + 1;
  issue_tile(0, 0);

  for (int i = 0; i < ntiles; ++i) {
    const int buf = i & 1;
    const int kv  = i * 64;
    if (i + 1 < ntiles) {
      issue_tile(buf ^ 1, kv + 64);                       // 8 more in flight
      asm volatile("s_wait_asynccnt 0x8" ::: "memory");   // current tile done
    } else {
      asm volatile("s_wait_asynccnt 0x0" ::: "memory");
    }
    __syncthreads();

    // scores S = Q @ K^T (Q pre-scaled) for 4 key sub-tiles of 16
    f32x8 s[4];
#pragma unroll
    for (int nt = 0; nt < 4; ++nt) {
      f32x8 z;
#pragma unroll
      for (int r = 0; r < 8; ++r) z[r] = 0.f;
      const bf16x16 kb0 = ld16(&Ks[buf][nt * 16 + ln][lh * 16]);
      const bf16x16 kb1 = ld16(&Ks[buf][nt * 16 + ln][32 + lh * 16]);
      z = __builtin_amdgcn_wmma_f32_16x16x32_bf16(false, qa0, false, kb0,
                                                  (short)0, z, false, false);
      z = __builtin_amdgcn_wmma_f32_16x16x32_bf16(false, qa1, false, kb1,
                                                  (short)0, z, false, false);
      s[nt] = z;
    }

    // causal mask is only active on the diagonal tile (kv+63 < q0 otherwise)
    if (i == ntiles - 1) {
#pragma unroll
      for (int nt = 0; nt < 4; ++nt) {
        const int kg = kv + nt * 16 + ln;
#pragma unroll
        for (int r = 0; r < 8; ++r) {
          const int qg = q0 + r + 8 * lh;
          if (kg > qg) s[nt][r] = -1e30f;
        }
      }
    }

    // row max (only remaining cross-lane reduction)
    float rmax[8];
#pragma unroll
    for (int r = 0; r < 8; ++r)
      rmax[r] = fmaxf(fmaxf(s[0][r], s[1][r]), fmaxf(s[2][r], s[3][r]));
#pragma unroll
    for (int r = 0; r < 8; ++r)
#pragma unroll
      for (int off = 8; off > 0; off >>= 1)
        rmax[r] = fmaxf(rmax[r], __shfl_xor(rmax[r], off, 16));

    // online softmax rescale: O and l share the same row layout
#pragma unroll
    for (int r = 0; r < 8; ++r) {
      const float mn = fmaxf(m_i[r], rmax[r]);
      const float al = __expf(m_i[r] - mn);
      m_i[r] = mn;
      l_acc[r] *= al;
#pragma unroll
      for (int nt = 0; nt < 4; ++nt) O[nt][r] *= al;
    }
#pragma unroll
    for (int nt = 0; nt < 4; ++nt)
#pragma unroll
      for (int r = 0; r < 8; ++r)
        s[nt][r] = __expf(s[nt][r] - m_i[r]);

    // C-layout -> A-layout re-staging of P through per-wave LDS
#pragma unroll
    for (int nt = 0; nt < 4; ++nt)
#pragma unroll
      for (int r = 0; r < 8; ++r)
        Pl[wv][r + 8 * lh][nt * 16 + ln] = f2bf(s[nt][r]);
    __syncthreads();

    // O += P @ V ; l += P @ ones (rowsum on the matrix unit)
#pragma unroll
    for (int kc = 0; kc < 2; ++kc) {
      const unsigned short* pb = &Pl[wv][ln][kc * 32];
      const bf16x16 pa = cat(ld8(pb + lh * 16), ld8(pb + lh * 16 + 8));
#pragma unroll
      for (int nt = 0; nt < 4; ++nt) {
        const bf16x16 vb = ld16(&Vt[buf][nt * 16 + ln][kc * 32 + lh * 16]);
        O[nt] = __builtin_amdgcn_wmma_f32_16x16x32_bf16(
            false, pa, false, vb, (short)0, O[nt], false, false);
      }
      l_acc = __builtin_amdgcn_wmma_f32_16x16x32_bf16(
          false, pa, false, onesb, (short)0, l_acc, false, false);
    }
    __syncthreads();
  }

  // normalize and store attention output (bf16) in [token][h*64+d] layout
#pragma unroll
  for (int nt = 0; nt < 4; ++nt)
#pragma unroll
    for (int r = 0; r < 8; ++r) {
      const float o = O[nt][r] / l_acc[r];
      AO[((size_t)b * SEQ + q0 + r + 8 * lh) * rowstr + headoff + nt * 16 + ln] =
          f2bf(o);
    }
}

// ---- launcher ------------------------------------------------------------

extern "C" void kernel_launch(void* const* d_in, const int* in_sizes, int n_in,
                              void* d_out, int out_size, void* d_ws, size_t ws_size,
                              hipStream_t stream) {
  (void)in_sizes; (void)n_in; (void)out_size; (void)ws_size;
  const float* X  = (const float*)d_in[0];
  // d_in[1] = mask (causal; encoded directly in the kernel)
  const float* Wq = (const float*)d_in[2];
  const float* bq = (const float*)d_in[3];
  const float* Wk = (const float*)d_in[4];
  const float* bk = (const float*)d_in[5];
  const float* Wv = (const float*)d_in[6];
  const float* bv = (const float*)d_in[7];
  const float* Wo = (const float*)d_in[8];
  const float* bo = (const float*)d_in[9];
  float* out = (float*)d_out;

  unsigned short* ws = (unsigned short*)d_ws;
  const size_t tokElems = (size_t)TOKENS * HIDDEN;   // 4096*1024
  const size_t wElems   = (size_t)HIDDEN * HIDDEN;   // 1024*1024
  unsigned short* Xb  = ws;                  // [tok][k] bf16
  unsigned short* Wtq = Xb  + tokElems;      // [n][k] bf16
  unsigned short* Wtk = Wtq + wElems;
  unsigned short* Wtv = Wtk + wElems;
  unsigned short* Wto = Wtv + wElems;
  unsigned short* Qb  = Wto + wElems;        // [tok][h*64+d] bf16 (pre-scaled)
  unsigned short* Kb  = Qb  + tokElems;
  unsigned short* Vb  = Kb  + tokElems;      // head-transposed: [b][h][d][s]
  unsigned short* AO  = Vb  + tokElems;

  // precision/layout prep (one-shot, bandwidth-trivial)
  cvt_bf16_kernel<<<dim3((unsigned)(tokElems / 2048)), dim3(256), 0, stream>>>(X, Xb);
  dim3 gt(HIDDEN / 32, HIDDEN / 32);
  transpose_cvt_kernel<<<gt, dim3(256), 0, stream>>>(Wq, Wtq, HIDDEN, HIDDEN);
  transpose_cvt_kernel<<<gt, dim3(256), 0, stream>>>(Wk, Wtk, HIDDEN, HIDDEN);
  transpose_cvt_kernel<<<gt, dim3(256), 0, stream>>>(Wv, Wtv, HIDDEN, HIDDEN);
  transpose_cvt_kernel<<<gt, dim3(256), 0, stream>>>(Wo, Wto, HIDDEN, HIDDEN);

  dim3 gg(HIDDEN / 128, TOKENS / 128);
  dim3 bg(256);
  gemm_bias_kernel<unsigned short, false><<<gg, bg, 0, stream>>>(
      Xb, Wtq, bq, Qb, TOKENS, HIDDEN, HIDDEN, 0.125f);   // fold 1/sqrt(d)
  gemm_bias_kernel<unsigned short, false><<<gg, bg, 0, stream>>>(
      Xb, Wtk, bk, Kb, TOKENS, HIDDEN, HIDDEN, 1.0f);
  gemm_bias_kernel<unsigned short, true><<<gg, bg, 0, stream>>>(
      Xb, Wtv, bv, Vb, TOKENS, HIDDEN, HIDDEN, 1.0f);

  dim3 ga(SEQ / 64, NUM_HEAD, BATCH);
  flash_attn_kernel<<<ga, dim3(128), 0, stream>>>(Qb, Kb, Vb, AO);

  gemm_bias_kernel<float, false><<<gg, bg, 0, stream>>>(
      AO, Wto, bo, out, TOKENS, HIDDEN, HIDDEN, 1.0f);
}
